// CausalSparseAttention_21715354648866
// MI455X (gfx1250) — compile-verified, matrix operand order checked by reading, not run on hardware
//
#include <hip/hip_runtime.h>

// ---------------------------------------------------------------------------
// Types
// ---------------------------------------------------------------------------
typedef __bf16 bf16;
typedef __attribute__((ext_vector_type(8)))  __bf16 v8bf;
typedef __attribute__((ext_vector_type(16))) __bf16 v16bf;
typedef __attribute__((ext_vector_type(8)))  float  v8f;

// Problem constants
#define SEQ_L   4096
#define DMODEL  1024
#define NHEADS  16
#define HDIM    64
#define WINDOW  512

// ds_swizzle group-of-32 XOR shuffle (wave32-safe; xor<16 stays inside the
// 16-lane row groups of the WMMA C layout).
#define SWZF(v, x) __builtin_bit_cast(float, __builtin_amdgcn_ds_swizzle( \
        __builtin_bit_cast(int, (float)(v)), (((x) << 10) | 0x1F)))

static __device__ inline v16bf mk16(v8bf lo, v8bf hi) {
    v16bf r;
#pragma unroll
    for (int i = 0; i < 8; ++i) { r[i] = lo[i]; r[8 + i] = hi[i]; }
    return r;
}

static __device__ inline v8f wmma_bf16(v16bf a, v16bf b, v8f c) {
    // D = A(16x32 bf16) * B(32x16 bf16) + C(16x16 f32)
    return __builtin_amdgcn_wmma_f32_16x16x32_bf16(
        /*neg_a=*/false, a, /*neg_b=*/false, b,
        /*c_mod=*/(short)0, c, /*reuse_a=*/false, /*reuse_b=*/false);
}

// Async global->LDS 16-byte copy (GLOBAL_LOAD_ASYNC_TO_LDS_B128, ASYNCcnt).
// Inline asm per CDNA5 ISA for toolchain portability.
static __device__ inline void async_copy16(void* lds_dst, const void* gsrc) {
    unsigned ldsa = (unsigned)(uintptr_t)lds_dst;   // low 32 bits = LDS offset
    asm volatile("global_load_async_to_lds_b128 %0, %1, off"
                 :: "v"(ldsa), "v"(gsrc) : "memory");
}

static __device__ inline void wait_async0() {
#if __has_builtin(__builtin_amdgcn_s_wait_asynccnt)
    __builtin_amdgcn_s_wait_asynccnt(0);
#else
    asm volatile("s_wait_asynccnt 0x0" ::: "memory");
#endif
}

// ---------------------------------------------------------------------------
// fp32 -> bf16 conversion (plain)
// ---------------------------------------------------------------------------
__global__ __launch_bounds__(256) void cvt_f32_bf16(const float* __restrict__ in,
                                                    bf16* __restrict__ out, int n) {
    int i = blockIdx.x * 256 + threadIdx.x;
    if (i < n) out[i] = (bf16)in[i];
}

// fp32 (1024 x N row-major) -> bf16 transposed (N x 1024 row-major).
// K is always DMODEL=1024 -> pure shift/mask indexing (no integer division).
__global__ __launch_bounds__(256) void cvt_f32_bf16_t(const float* __restrict__ in,
                                                      bf16* __restrict__ out,
                                                      int N) {
    int i = blockIdx.x * 256 + threadIdx.x;   // index into out (N*1024)
    if (i >= N * DMODEL) return;
    int n = i >> 10;            // / DMODEL
    int k = i & (DMODEL - 1);   // % DMODEL
    out[i] = (bf16)in[(size_t)k * N + n];
}

// ---------------------------------------------------------------------------
// RoPE (in place on head-major [h][pos][64] bf16 Q and K)
// ---------------------------------------------------------------------------
__global__ __launch_bounds__(256) void rope_kernel(bf16* __restrict__ q,
                                                   bf16* __restrict__ k) {
    int i = blockIdx.x * 256 + threadIdx.x;       // NHEADS*SEQ_L*(HDIM/2)
    if (i >= NHEADS * SEQ_L * (HDIM / 2)) return;
    int j   = i & (HDIM / 2 - 1);
    int pos = (i >> 5) & (SEQ_L - 1);
    int h   = i >> 17;                            // / (SEQ_L * 32)
    // 10000^(-2j/64) = exp2(-2j * log2(10000)/64); single transcendental.
    const float c0 = 13.287712379549449f / (float)HDIM;  // log2(10000)/64
    float inv_freq = exp2f(-(float)(2 * j) * c0);
    float s, c;
    __sincosf((float)pos * inv_freq, &s, &c);
    int base = ((h * SEQ_L) + pos) * HDIM + 2 * j;
    float q0 = (float)q[base], q1 = (float)q[base + 1];
    q[base]     = (bf16)(q0 * c - q1 * s);
    q[base + 1] = (bf16)(q1 * c + q0 * s);
    float k0 = (float)k[base], k1 = (float)k[base + 1];
    k[base]     = (bf16)(k0 * c - k1 * s);
    k[base + 1] = (bf16)(k1 * c + k0 * s);
}

// ---------------------------------------------------------------------------
// Tiled bf16 WMMA GEMM: C(MxN) = A(MxK) * Bt^T, where Bt is B transposed
// (N x K row-major).  Block tile 128x128, 256 threads = 8 waves (2x4),
// wave tile 64x32.  Both A and Bt tiles staged with async global->LDS b128.
// MODE 0: scatter into head-major q/k (and dim-major v) bf16 buffers.
// MODE 1: write fp32 row-major into cf (output projection).
// ---------------------------------------------------------------------------
template <int MODE>
__global__ __launch_bounds__(256) void gemm_bf16_wmma(
    const bf16* __restrict__ A, const bf16* __restrict__ Bt,
    int M, int N, int K,
    bf16* __restrict__ qh, bf16* __restrict__ kh, bf16* __restrict__ vh,
    float* __restrict__ cf) {

    const int tid  = threadIdx.x;
    const int lane = tid & 31;
    const int wv   = tid >> 5;
    const int lh   = lane & 15;       // row/col within 16
    const int lq   = lane >> 4;       // half-wave select
    const int wr   = wv >> 2;         // wave row (0..1) -> 64 rows each
    const int wc   = wv & 3;          // wave col (0..3) -> 32 cols each
    const int m0   = blockIdx.y * 128;
    const int n0   = blockIdx.x * 128;

    __shared__ bf16 As[128 * 40];     // [row][k] stride 40 (pad, 16B aligned)
    __shared__ bf16 Bs[128 * 40];     // [col][k] stride 40

    v8f acc[4][2] = {};

    // Marching global source pointers for the async tile copies.
    const int cprow = tid >> 2;              // 0..63   (rows 0..63 | 64..127)
    const int cpqd  = tid & 3;               // 16B chunk within a 32-elem row
    const bf16* ap0 = A  + (size_t)(m0 + cprow)      * K + cpqd * 8;
    const bf16* ap1 = A  + (size_t)(m0 + cprow + 64) * K + cpqd * 8;
    const bf16* bp0 = Bt + (size_t)(n0 + cprow)      * K + cpqd * 8;
    const bf16* bp1 = Bt + (size_t)(n0 + cprow + 64) * K + cpqd * 8;
    bf16* lA0 = &As[cprow * 40 + cpqd * 8];
    bf16* lA1 = &As[(cprow + 64) * 40 + cpqd * 8];
    bf16* lB0 = &Bs[cprow * 40 + cpqd * 8];
    bf16* lB1 = &Bs[(cprow + 64) * 40 + cpqd * 8];

    for (int k0 = 0; k0 < K; k0 += 32) {
        __syncthreads();
        // --- async stage A tile (128x32) and Bt tile (128x32), 16B chunks ---
        async_copy16(lA0, ap0);
        async_copy16(lA1, ap1);
        async_copy16(lB0, bp0);
        async_copy16(lB1, bp1);
        ap0 += 32; ap1 += 32; bp0 += 32; bp1 += 32;
        wait_async0();
        __syncthreads();

        // --- fragments per ISA layout: lane lh = row/col, chunks at lq*8, +16
        v16bf af[4];
#pragma unroll
        for (int mi = 0; mi < 4; ++mi) {
            int row = wr * 64 + mi * 16 + lh;
            v8bf lo = *reinterpret_cast<const v8bf*>(&As[row * 40 + lq * 8]);
            v8bf hi = *reinterpret_cast<const v8bf*>(&As[row * 40 + lq * 8 + 16]);
            af[mi] = mk16(lo, hi);
        }
        v16bf bfm[2];
#pragma unroll
        for (int ni = 0; ni < 2; ++ni) {
            int col = wc * 32 + ni * 16 + lh;
            v8bf lo = *reinterpret_cast<const v8bf*>(&Bs[col * 40 + lq * 8]);
            v8bf hi = *reinterpret_cast<const v8bf*>(&Bs[col * 40 + lq * 8 + 16]);
            bfm[ni] = mk16(lo, hi);
        }
#pragma unroll
        for (int mi = 0; mi < 4; ++mi)
#pragma unroll
            for (int ni = 0; ni < 2; ++ni)
                acc[mi][ni] = wmma_bf16(af[mi], bfm[ni], acc[mi][ni]);
    }

    // --- epilogue: C layout lane lh = column, vgpr r = row r + lq*8.
    //     One base + constant stride per 16x16 subtile -> immediate offsets.
#pragma unroll
    for (int mi = 0; mi < 4; ++mi) {
#pragma unroll
        for (int ni = 0; ni < 2; ++ni) {
            const int n  = n0 + wc * 32 + ni * 16 + lh;
            const int mb = m0 + wr * 64 + mi * 16 + lq * 8;
            if constexpr (MODE == 0) {
                // n = sel*1024 + head*64 + dim
                const int sel = n >> 10;
                const int hh  = (n >> 6) & 15;
                const int d   = n & 63;
                bf16*  dst;
                size_t base;
                size_t stride;
                if (sel == 0) {        // Q: [h][pos][d]
                    dst = qh; base = ((size_t)(hh << 12) + mb) * HDIM + d; stride = HDIM;
                } else if (sel == 1) { // K: [h][pos][d]
                    dst = kh; base = ((size_t)(hh << 12) + mb) * HDIM + d; stride = HDIM;
                } else {               // V: dim-major [h][d][pos]
                    dst = vh; base = (((size_t)(hh << 6) + d) << 12) + mb; stride = 1;
                }
#pragma unroll
                for (int r = 0; r < 8; ++r)
                    dst[base + (size_t)r * stride] = (bf16)acc[mi][ni][r];
            } else {
                float* dst = cf + (size_t)mb * N + n;
#pragma unroll
                for (int r = 0; r < 8; ++r)
                    dst[(size_t)r * N] = acc[mi][ni][r];
            }
        }
    }
}

// ---------------------------------------------------------------------------
// Sliding-window flash attention.
// Grid: (SEQ_L/64, NHEADS). Block: 128 threads = 4 waves, 16 queries each.
// Q/K head-major [h][pos][64]: K fragments load straight from global (the
// B [N][K] layout IS the memory layout).  V dim-major [h][d][pos]: P*V
// B-fragments also load straight from global.  The only LDS use is the
// per-wave P relayout (C layout -> A layout), ordered by the wave's
// in-order DS guarantee -- no workgroup barriers.  All inner-loop accesses
// are marching-base-pointer + immediate offset.
// ---------------------------------------------------------------------------
__global__ __launch_bounds__(128) void attn_kernel(
    const bf16* __restrict__ qh, const bf16* __restrict__ kh,
    const bf16* __restrict__ vh, bf16* __restrict__ ob) {

    const int h    = blockIdx.y;
    const int q0   = blockIdx.x * 64;
    const int tid  = threadIdx.x;
    const int lane = tid & 31;
    const int wv   = tid >> 5;        // wave 0..3 -> 16 queries each
    const int lh   = lane & 15;
    const int lq   = lane >> 4;

    __shared__ bf16 Pb[4][16 * 40];   // per-wave P tile [row][key] stride 40

    // Q fragments for this wave's 16 rows (loaded once, A layout)
    const int qrow = q0 + wv * 16 + lh;
    const size_t qbase = ((size_t)(h << 12) + qrow) * HDIM + lq * 8;
    v16bf aq0 = mk16(*reinterpret_cast<const v8bf*>(qh + qbase),
                     *reinterpret_cast<const v8bf*>(qh + qbase + 16));
    v16bf aq1 = mk16(*reinterpret_cast<const v8bf*>(qh + qbase + 32),
                     *reinterpret_cast<const v8bf*>(qh + qbase + 48));

    float mrow[8], lrow[8];
#pragma unroll
    for (int r = 0; r < 8; ++r) { mrow[r] = -1e30f; lrow[r] = 0.0f; }
    v8f o[4] = {};

    const int kstart = (q0 > (WINDOW - 1)) ? ((q0 - (WINDOW - 1)) & ~31) : 0;
    const int kend   = q0 + 64;

    // Marching base pointers (advance by constants; loads use imm offsets).
    const bf16* kp  = kh + ((size_t)(h << 12) + kstart + lh) * HDIM + lq * 8;
    const bf16* vp  = vh + (((size_t)(h << 6) + lh) << 12) + kstart + lq * 8;
    const bf16* kpf = kh + ((size_t)(h << 12) + kstart + lane) * HDIM;
    const bf16* vpf = vh + (((size_t)(h << 6) + (lane << 1)) << 12) + kstart;
    // per-wave LDS P pointers
    bf16* pw = &Pb[wv][0];
    const bf16* pr = &Pb[wv][lh * 40 + lq * 8];

    for (int kt = kstart; kt < kend; kt += 32) {
        // WGP-scope prefetch of the next key/value tiles
        if (kt + 32 < kend) {
            __builtin_prefetch(kpf + 32 * HDIM, 0, 3);
            __builtin_prefetch(vpf + 32, 0, 3);
        }

        // ---- S = Q * K^T (scores), 2 key sub-tiles of 16 ----
        v8f s[2] = {};
#pragma unroll
        for (int ni = 0; ni < 2; ++ni) {
            const bf16* kb = kp + ni * (16 * HDIM);
            v16bf b0 = mk16(*reinterpret_cast<const v8bf*>(kb),
                            *reinterpret_cast<const v8bf*>(kb + 16));
            v16bf b1 = mk16(*reinterpret_cast<const v8bf*>(kb + 32),
                            *reinterpret_cast<const v8bf*>(kb + 48));
            s[ni] = wmma_bf16(aq0, b0, s[ni]);
            s[ni] = wmma_bf16(aq1, b1, s[ni]);
        }

        // ---- online softmax per row (row = r + lq*8, cols across 16 lanes)
        const int kc0 = kt + lh;
        const int kc1 = kt + 16 + lh;
#pragma unroll
        for (int r = 0; r < 8; ++r) {
            const int Mr = r + lq * 8;
            const int qi = q0 + wv * 16 + Mr;
            float x0 = s[0][r] * 0.125f;
            float x1 = s[1][r] * 0.125f;
            if (!(qi >= kc0 && (qi - kc0) < WINDOW)) x0 = -1e30f;
            if (!(qi >= kc1 && (qi - kc1) < WINDOW)) x1 = -1e30f;
            float tmax = fmaxf(x0, x1);
            tmax = fmaxf(tmax, SWZF(tmax, 1));
            tmax = fmaxf(tmax, SWZF(tmax, 2));
            tmax = fmaxf(tmax, SWZF(tmax, 4));
            tmax = fmaxf(tmax, SWZF(tmax, 8));
            float nm = fmaxf(mrow[r], tmax);
            float sc = __expf(mrow[r] - nm);
            mrow[r] = nm;
            float p0 = __expf(x0 - nm);
            float p1 = __expf(x1 - nm);
            float ts = p0 + p1;
            ts += SWZF(ts, 1);
            ts += SWZF(ts, 2);
            ts += SWZF(ts, 4);
            ts += SWZF(ts, 8);
            lrow[r] = lrow[r] * sc + ts;
#pragma unroll
            for (int n = 0; n < 4; ++n) o[n][r] *= sc;
            pw[Mr * 40 + lh]      = (bf16)p0;
            pw[Mr * 40 + 16 + lh] = (bf16)p1;
        }

        // ---- O += P * V : A = P (16x32 keys, via per-wave LDS relayout),
        //      B = V dim-major from global (32 keys x 16 dims) ----
        v16bf aP = mk16(*reinterpret_cast<const v8bf*>(pr),
                        *reinterpret_cast<const v8bf*>(pr + 16));
#pragma unroll
        for (int n = 0; n < 4; ++n) {
            const bf16* vb = vp + (size_t)n * (16 * SEQ_L);
            v16bf bV = mk16(*reinterpret_cast<const v8bf*>(vb),
                            *reinterpret_cast<const v8bf*>(vb + 16));
            o[n] = wmma_bf16(aP, bV, o[n]);
        }

        kp  += 32 * HDIM;
        vp  += 32;
        kpf += 32 * HDIM;
        vpf += 32;
    }

    // ---- normalize and store row-major [pos][h*64+d] bf16 ----
#pragma unroll
    for (int r = 0; r < 8; ++r) {
        const int Mr = r + lq * 8;
        const int qi = q0 + wv * 16 + Mr;
        float inv = (lrow[r] > 0.0f) ? (1.0f / lrow[r]) : 0.0f;
#pragma unroll
        for (int n = 0; n < 4; ++n)
            ob[(size_t)qi * DMODEL + (h << 6) + n * 16 + lh] = (bf16)(o[n][r] * inv);
    }
}

// ---------------------------------------------------------------------------
// Launch
// ---------------------------------------------------------------------------
extern "C" void kernel_launch(void* const* d_in, const int* in_sizes, int n_in,
                              void* d_out, int out_size, void* d_ws, size_t ws_size,
                              hipStream_t stream) {
    const float* x     = (const float*)d_in[0];   // 1 x 4096 x 1024
    const float* w_qkv = (const float*)d_in[1];   // 1024 x 3072
    const float* w_out = (const float*)d_in[2];   // 1024 x 1024
    float* out = (float*)d_out;                   // 1 x 4096 x 1024 fp32

    char* ws = (char*)d_ws;
    bf16* xb    = (bf16*)ws;  ws += (size_t)SEQ_L * DMODEL * 2;       // 8 MB
    bf16* wqkvt = (bf16*)ws;  ws += (size_t)DMODEL * 3 * DMODEL * 2;  // 6 MB (transposed)
    bf16* woutt = (bf16*)ws;  ws += (size_t)DMODEL * DMODEL * 2;      // 2 MB (transposed)
    bf16* qhb   = (bf16*)ws;  ws += (size_t)NHEADS * SEQ_L * HDIM * 2;
    bf16* khb   = (bf16*)ws;  ws += (size_t)NHEADS * SEQ_L * HDIM * 2;
    bf16* vhb   = (bf16*)ws;  ws += (size_t)NHEADS * SEQ_L * HDIM * 2;
    bf16* obb   = (bf16*)ws;  ws += (size_t)SEQ_L * DMODEL * 2;       // ~50 MB total

    // 1) fp32 -> bf16 conversions (weights transposed for async tile staging)
    {
        int n1 = SEQ_L * DMODEL;
        cvt_f32_bf16<<<(n1 + 255) / 256, 256, 0, stream>>>(x, xb, n1);
        int n2 = DMODEL * 3 * DMODEL;
        cvt_f32_bf16_t<<<(n2 + 255) / 256, 256, 0, stream>>>(w_qkv, wqkvt, 3 * DMODEL);
        int n3 = DMODEL * DMODEL;
        cvt_f32_bf16_t<<<(n3 + 255) / 256, 256, 0, stream>>>(w_out, woutt, DMODEL);
    }
    // 2) QKV projection, scatter into head-major q/k + dim-major v
    gemm_bf16_wmma<0><<<dim3((3 * DMODEL) / 128, SEQ_L / 128), 256, 0, stream>>>(
        xb, wqkvt, SEQ_L, 3 * DMODEL, DMODEL, qhb, khb, vhb, nullptr);
    // 3) RoPE on Q, K
    {
        int n = NHEADS * SEQ_L * (HDIM / 2);
        rope_kernel<<<(n + 255) / 256, 256, 0, stream>>>(qhb, khb);
    }
    // 4) sliding-window flash attention (barrier-free)
    attn_kernel<<<dim3(SEQ_L / 64, NHEADS), 128, 0, stream>>>(qhb, khb, vhb, obb);
    // 5) output projection -> fp32 d_out
    gemm_bf16_wmma<1><<<dim3(DMODEL / 128, SEQ_L / 128), 256, 0, stream>>>(
        obb, woutt, SEQ_L, DMODEL, DMODEL, nullptr, nullptr, nullptr, out);
}
